// Ulstm_Conv_Former_5970004541673
// MI455X (gfx1250) — matrix operand order repaired and checked by reading, not compiled, not run on hardware
//
#include <hip/hip_runtime.h>

// ---------------------------------------------------------------------------
// ConvFormer forward for MI455X (gfx1250): f16 WMMA implicit-GEMM convs,
// fused attention (pair tensor never materialized), f32 accumulate.
// ---------------------------------------------------------------------------

typedef __attribute__((ext_vector_type(16))) _Float16 v16h;
typedef __attribute__((ext_vector_type(8)))  _Float16 v8h;
typedef __attribute__((ext_vector_type(8)))  float    v8f;

#define T_ 4
#define C_ 64
#define S_ 32768            // 32*32*32 voxels per frame
#define NCONVPK 110592      // 27*2*4*32*16 packed halves per conv weight
#define NMMPK 4096          // 2*4*32*16 packed halves per 64x64 matmul weight

union AFrag { v16h v; v8h h[2]; };

static __device__ inline v8h v8h_zero() {
  v8h z;
#pragma unroll
  for (int i = 0; i < 8; ++i) z[i] = (_Float16)0.f;
  return z;
}

static __device__ inline v8h relu8(v8h x) {
  v8h r;
#pragma unroll
  for (int i = 0; i < 8; ++i) { _Float16 v = x[i]; r[i] = v > (_Float16)0.f ? v : (_Float16)0.f; }
  return r;
}

// ---------------------------------------------------------------------------
// K0: pack weights into per-lane WMMA B-fragment order.
// Conv pack idx  = (((tap*2+kc)*4+n)*32+lane)*16+j ; co=16n+lane%16, ci=32kc+(j/8)*16+(lane/16)*8+j%8
// Matmul pack idx = ((kc*4+n)*32+lane)*16+j        ; o =16n+lane%16, i =32kc+(j/8)*16+(lane/16)*8+j%8
// ---------------------------------------------------------------------------
__global__ __launch_bounds__(256)
void pack_weights(const float* __restrict__ Wq, const float* __restrict__ Wk,
                  const float* __restrict__ Wv, const float* __restrict__ Wf,
                  const float* __restrict__ W1, const float* __restrict__ W2,
                  const float* __restrict__ W3,
                  _Float16* __restrict__ pq, _Float16* __restrict__ pk,
                  _Float16* __restrict__ pv, _Float16* __restrict__ pf,
                  _Float16* __restrict__ p1, _Float16* __restrict__ p2,
                  _Float16* __restrict__ p3)
{
  const int gid = blockIdx.x * 256 + threadIdx.x;
  if (gid < 4 * NCONVPK) {
    const int wi = gid / NCONVPK;
    const int r  = gid % NCONVPK;
    const int j = r & 15, lane = (r >> 4) & 31, n = (r >> 9) & 3;
    const int kc = (r >> 11) & 1, tap = r >> 12;          // tap = (dz*3+dy)*3+dx
    const int co = 16 * n + (lane & 15);
    const int ci = 32 * kc + (j >> 3) * 16 + (lane >> 4) * 8 + (j & 7);
    const float* W = (wi == 0) ? Wq : (wi == 1) ? Wk : (wi == 2) ? Wv : Wf;
    _Float16*    P = (wi == 0) ? pq : (wi == 1) ? pk : (wi == 2) ? pv : pf;
    P[r] = (_Float16)W[(co * 64 + ci) * 27 + tap];
  } else {
    const int g = gid - 4 * NCONVPK;
    if (g < 3 * NMMPK) {
      const int wm = g / NMMPK;
      const int r  = g % NMMPK;
      const int j = r & 15, lane = (r >> 4) & 31, n = (r >> 9) & 3, kc = r >> 11;
      const int o = 16 * n + (lane & 15);
      const int i = 32 * kc + (j >> 3) * 16 + (lane >> 4) * 8 + (j & 7);
      const float* W = (wm == 0) ? W1 : (wm == 1) ? W2 : W3;
      _Float16*    P = (wm == 0) ? p1 : (wm == 1) ? p2 : p3;
      P[r] = (_Float16)W[o * 64 + i];
    }
  }
}

// ---------------------------------------------------------------------------
// K1: (e|c) + pos -> f16, channel-last [t][s][c]  (coalesced 2B writes)
// ---------------------------------------------------------------------------
__global__ __launch_bounds__(256)
void add_pos_pack(const float* __restrict__ ein, const float* __restrict__ cin,
                  const float* __restrict__ pos,
                  _Float16* __restrict__ eh, _Float16* __restrict__ ch)
{
  const int which = blockIdx.y;
  const float* src = which ? cin : ein;
  _Float16*    dst = which ? ch : eh;
  const size_t gid = (size_t)blockIdx.x * 256 + threadIdx.x;  // output order index
  const int c = (int)(gid & 63);
  const size_t rest = gid >> 6;
  const int s = (int)(rest & (S_ - 1));
  const int t = (int)(rest >> 15);
  const float v = src[((size_t)t * 64 + c) * S_ + s] + pos[(size_t)c * S_ + s];
  dst[gid] = (_Float16)v;
}

// ---------------------------------------------------------------------------
// K2: 3x3x3 SAME conv, 64->64, implicit GEMM via v_wmma_f32_16x16x32_f16.
// Block = one (t,z,y) row: 32 voxels x 64 out-ch. 8 waves, wave(m,n) owns a
// 16x16 tile. Per dz: stage 3 halo'd y-rows into LDS, then 3*3*2 WMMAs/wave.
// ---------------------------------------------------------------------------
__global__ __launch_bounds__(256)
void conv3d_wmma(const _Float16* __restrict__ in,   // [T][S][64] f16
                 const _Float16* __restrict__ wpk,  // packed B fragments
                 const float* __restrict__ bias,
                 float* __restrict__ outp)          // [T][S][64] f32 (pre-norm)
{
  __shared__ _Float16 sm[3 * 34 * 64];              // 13056 B

  const int bx = blockIdx.x;                        // t*1024 + z*32 + y
  const int t = bx >> 10;
  const int z = (bx >> 5) & 31;
  const int y = bx & 31;

  const int tid = threadIdx.x;
  const int w  = tid >> 5;
  const int l  = tid & 31;
  const int m  = w & 1;                             // x tile
  const int n  = w >> 1;                            // co tile
  const int lm = l & 15;
  const int lh = l >> 4;

  v8f acc = {0.f, 0.f, 0.f, 0.f, 0.f, 0.f, 0.f, 0.f};

  for (int dz = 0; dz < 3; ++dz) {
    __syncthreads();
    const int zsrc = z + dz - 1;
    for (int chn = tid; chn < 3 * 34 * 8; chn += 256) {
      const int part = chn & 7;                     // 8-half chunk within channels
      const int pair = chn >> 3;                    // (dy row, xx)
      const int xx   = pair % 34;
      const int r    = pair / 34;
      const int ysrc = y + r - 1;
      const int xsrc = xx - 1;
      v8h val = v8h_zero();
      if (zsrc >= 0 && zsrc < 32 && ysrc >= 0 && ysrc < 32 && xsrc >= 0 && xsrc < 32) {
        const int s = (zsrc * 32 + ysrc) * 32 + xsrc;
        val = *(const v8h*)(in + ((size_t)t * S_ + s) * C_ + part * 8);
      }
      *(v8h*)(sm + pair * 64 + part * 8) = val;
    }
    __syncthreads();

#pragma unroll
    for (int dy = 0; dy < 3; ++dy) {
#pragma unroll
      for (int dx = 0; dx < 3; ++dx) {
        const int tap = (dz * 3 + dy) * 3 + dx;
        const int xx  = 16 * m + lm + dx;           // 0..33 (halo'd)
#pragma unroll
        for (int kc = 0; kc < 2; ++kc) {
          AFrag a;
          const int base = (dy * 34 + xx) * 64 + kc * 32 + lh * 8;
          a.h[0] = *(const v8h*)(sm + base);        // K = lh*8 + 0..7
          a.h[1] = *(const v8h*)(sm + base + 16);   // K = 16 + lh*8 + 0..7
          const v16h b = *(const v16h*)(wpk + (((tap * 2 + kc) * 4 + n) * 32 + l) * 16);
          acc = __builtin_amdgcn_wmma_f32_16x16x32_f16(false, a.v, false, b,
                                                       (short)0, acc, false, false);
        }
      }
    }
  }

  const int co = 16 * n + lm;
  const float bv = bias[co];
#pragma unroll
  for (int r = 0; r < 8; ++r) {
    const int x = 16 * m + r + lh * 8;              // C layout: M = r + 8*(lane/16)
    const int s = (z * 32 + y) * 32 + x;
    outp[((size_t)t * S_ + s) * C_ + co] = acc[r] + bv;
  }
}

// ---------------------------------------------------------------------------
// K3: InstanceNorm (biased var, eps 1e-5) + LeakyReLU(0.2).
// Block per (t,c). mode 0: f16 channel-last out ; mode 1: f32 channel-first.
// ---------------------------------------------------------------------------
__global__ __launch_bounds__(256)
void inorm_lrelu(const float* __restrict__ src,     // [T][S][64]
                 _Float16* __restrict__ dsth,       // [T][S][64]   (mode 0)
                 float* __restrict__ dstf,          // [T][64][S]   (mode 1)
                 int mode)
{
  __shared__ float ssum[256];
  __shared__ float ssq[256];
  const int tc = blockIdx.x;
  const int t = tc >> 6;
  const int c = tc & 63;
  const int tid = threadIdx.x;
  const float* base = src + (size_t)t * S_ * C_ + c;

  float s0 = 0.f, s1 = 0.f;
  for (int s = tid; s < S_; s += 256) {
    const float x = base[(size_t)s * C_];
    s0 += x; s1 += x * x;
  }
  ssum[tid] = s0; ssq[tid] = s1;
  __syncthreads();
  for (int off = 128; off > 0; off >>= 1) {
    if (tid < off) { ssum[tid] += ssum[tid + off]; ssq[tid] += ssq[tid + off]; }
    __syncthreads();
  }
  const float mean = ssum[0] * (1.f / S_);
  const float var  = ssq[0] * (1.f / S_) - mean * mean;
  const float rs   = rsqrtf(var + 1e-5f);

  for (int s = tid; s < S_; s += 256) {
    float x = (base[(size_t)s * C_] - mean) * rs;
    x = x > 0.f ? x : 0.2f * x;
    if (mode == 0) dsth[((size_t)t * S_ + s) * C_ + c] = (_Float16)x;
    else           dstf[((size_t)(t * 64 + c)) * S_ + s] = x;
  }
}

// ---------------------------------------------------------------------------
// K4: 1x1x1 channel matmuls  a = W1 q + b1 (y=0),  bkk = W2 k + b2 (y=1).
// ---------------------------------------------------------------------------
__global__ __launch_bounds__(256)
void chan_gemm(const _Float16* __restrict__ qh, const _Float16* __restrict__ kh,
               const _Float16* __restrict__ w1p, const _Float16* __restrict__ w2p,
               const float* __restrict__ bias1, const float* __restrict__ bias2,
               _Float16* __restrict__ ah, _Float16* __restrict__ bkh)
{
  const int which = blockIdx.y;
  const _Float16* src = which ? kh : qh;
  const _Float16* wp  = which ? w2p : w1p;
  const float*    bb  = which ? bias2 : bias1;
  _Float16*       dst = which ? bkh : ah;

  const int tid = threadIdx.x, w = tid >> 5, l = tid & 31;
  const int m = w & 1, n = w >> 1, lm = l & 15, lh = l >> 4;
  const int rowbase = blockIdx.x * 32 + 16 * m;     // rows over T_*S_
  const size_t arow = (size_t)(rowbase + lm) * C_;

  v8f acc = {0.f, 0.f, 0.f, 0.f, 0.f, 0.f, 0.f, 0.f};
#pragma unroll
  for (int kc = 0; kc < 2; ++kc) {
    AFrag a;
    a.h[0] = *(const v8h*)(src + arow + kc * 32 + lh * 8);
    a.h[1] = *(const v8h*)(src + arow + kc * 32 + lh * 8 + 16);
    const v16h b = *(const v16h*)(wp + ((kc * 4 + n) * 32 + l) * 16);
    acc = __builtin_amdgcn_wmma_f32_16x16x32_f16(false, a.v, false, b,
                                                 (short)0, acc, false, false);
  }
  const int co = 16 * n + lm;
  const float bv = bb[co];
#pragma unroll
  for (int r = 0; r < 8; ++r)
    dst[(size_t)(rowbase + r + lh * 8) * C_ + co] = (_Float16)(acc[r] + bv);
}

// ---------------------------------------------------------------------------
// K5: fused attention. Per (tq, 32-voxel tile):
//   agg = sum_tk sigmoid(W3 relu(a[tq]+bkk[tk]) + b3) * v[tk]   (pair in regs)
//   emit f16 (agg + q) for the final conv.
// ---------------------------------------------------------------------------
__global__ __launch_bounds__(256)
void attn_fuse(const _Float16* __restrict__ ah,     // [T][S][64]
               const _Float16* __restrict__ bkh,    // [T][S][64]
               const _Float16* __restrict__ vh,     // [T][S][64]
               const _Float16* __restrict__ qh,     // [T][S][64]
               const _Float16* __restrict__ w3p,
               const float* __restrict__ bias3,
               _Float16* __restrict__ aggq)         // [T][S][64]
{
  const int tq = blockIdx.y;
  const int vb = blockIdx.x * 32;
  const int tid = threadIdx.x, w = tid >> 5, l = tid & 31;
  const int m = w & 1, n = w >> 1, lm = l & 15, lh = l >> 4;
  const int co = 16 * n + lm;
  const float b3v = bias3[co];

  const size_t arow = ((size_t)tq * S_ + vb + 16 * m + lm) * C_;
  const v8h a0lo = *(const v8h*)(ah + arow + 0  + lh * 8);
  const v8h a0hi = *(const v8h*)(ah + arow + 16 + lh * 8);
  const v8h a1lo = *(const v8h*)(ah + arow + 32 + lh * 8);
  const v8h a1hi = *(const v8h*)(ah + arow + 48 + lh * 8);
  const v16h wb0 = *(const v16h*)(w3p + ((0 * 4 + n) * 32 + l) * 16);
  const v16h wb1 = *(const v16h*)(w3p + ((1 * 4 + n) * 32 + l) * 16);

  float agg[8] = {0.f, 0.f, 0.f, 0.f, 0.f, 0.f, 0.f, 0.f};

  for (int tk = 0; tk < T_; ++tk) {
    const size_t krow = ((size_t)tk * S_ + vb + 16 * m + lm) * C_;
    v8f acc = {0.f, 0.f, 0.f, 0.f, 0.f, 0.f, 0.f, 0.f};
    AFrag a;
    a.h[0] = relu8(a0lo + *(const v8h*)(bkh + krow + 0  + lh * 8));
    a.h[1] = relu8(a0hi + *(const v8h*)(bkh + krow + 16 + lh * 8));
    acc = __builtin_amdgcn_wmma_f32_16x16x32_f16(false, a.v, false, wb0,
                                                 (short)0, acc, false, false);
    a.h[0] = relu8(a1lo + *(const v8h*)(bkh + krow + 32 + lh * 8));
    a.h[1] = relu8(a1hi + *(const v8h*)(bkh + krow + 48 + lh * 8));
    acc = __builtin_amdgcn_wmma_f32_16x16x32_f16(false, a.v, false, wb1,
                                                 (short)0, acc, false, false);
#pragma unroll
    for (int r = 0; r < 8; ++r) {
      const float sc = 1.f / (1.f + __expf(-(acc[r] + b3v)));
      const int vox = vb + 16 * m + r + lh * 8;
      const float vv = (float)vh[((size_t)tk * S_ + vox) * C_ + co];
      agg[r] += sc * vv;
    }
  }
#pragma unroll
  for (int r = 0; r < 8; ++r) {
    const int vox = vb + 16 * m + r + lh * 8;
    const size_t o = ((size_t)tq * S_ + vox) * C_ + co;
    aggq[o] = (_Float16)(agg[r] + (float)qh[o]);
  }
}

// ---------------------------------------------------------------------------
extern "C" void kernel_launch(void* const* d_in, const int* in_sizes, int n_in,
                              void* d_out, int out_size, void* d_ws, size_t ws_size,
                              hipStream_t stream)
{
  (void)in_sizes; (void)n_in; (void)out_size; (void)ws_size;

  const float* c_in = (const float*)d_in[0];
  const float* e_in = (const float*)d_in[1];
  const float* pos  = (const float*)d_in[2];
  const float* Wq = (const float*)d_in[3];  const float* bq = (const float*)d_in[4];
  const float* Wk = (const float*)d_in[5];  const float* bk = (const float*)d_in[6];
  const float* Wv = (const float*)d_in[7];  const float* bv = (const float*)d_in[8];
  const float* W1 = (const float*)d_in[9];  const float* b1 = (const float*)d_in[10];
  const float* W2 = (const float*)d_in[11]; const float* b2 = (const float*)d_in[12];
  const float* W3 = (const float*)d_in[13]; const float* b3 = (const float*)d_in[14];
  const float* Wf = (const float*)d_in[15]; const float* bf = (const float*)d_in[16];
  float* out = (float*)d_out;

  char* ws = (char*)d_ws;
  const size_t HBUF = (size_t)T_ * S_ * C_ * sizeof(_Float16);   // 16 MiB
  _Float16* ep  = (_Float16*)(ws + 0 * HBUF);   // e + pos, f16 channel-last
  _Float16* cp  = (_Float16*)(ws + 1 * HBUF);   // c + pos
  _Float16* qh  = (_Float16*)(ws + 2 * HBUF);   // q (post-norm)
  _Float16* kh  = (_Float16*)(ws + 3 * HBUF);   // k
  _Float16* vh  = (_Float16*)(ws + 4 * HBUF);   // v
  _Float16* ah  = (_Float16*)(ws + 5 * HBUF);   // W1 q + b1
  _Float16* bkh = (_Float16*)(ws + 6 * HBUF);   // W2 k + b2
  _Float16* gq  = (_Float16*)(ws + 7 * HBUF);   // agg + q
  float*    tmp = (float*)   (ws + 8 * HBUF);   // pre-norm conv out, f32 (32 MiB)
  char* pk0 = ws + 8 * HBUF + (size_t)T_ * S_ * C_ * sizeof(float);
  _Float16* pwq = (_Float16*)(pk0);
  _Float16* pwk = (_Float16*)(pk0 + 1 * (size_t)NCONVPK * 2);
  _Float16* pwv = (_Float16*)(pk0 + 2 * (size_t)NCONVPK * 2);
  _Float16* pwf = (_Float16*)(pk0 + 3 * (size_t)NCONVPK * 2);
  _Float16* pw1 = (_Float16*)(pk0 + 4 * (size_t)NCONVPK * 2);
  _Float16* pw2 = (_Float16*)(pk0 + 4 * (size_t)NCONVPK * 2 + NMMPK * 2);
  _Float16* pw3 = (_Float16*)(pk0 + 4 * (size_t)NCONVPK * 2 + 2 * NMMPK * 2);

  const int packN = 4 * NCONVPK + 3 * NMMPK;
  pack_weights<<<(packN + 255) / 256, 256, 0, stream>>>(
      Wq, Wk, Wv, Wf, W1, W2, W3, pwq, pwk, pwv, pwf, pw1, pw2, pw3);

  add_pos_pack<<<dim3((T_ * S_ * C_) / 256, 2), 256, 0, stream>>>(e_in, c_in, pos, ep, cp);

  // q = ConvBlock(e+pos, Wq)
  conv3d_wmma<<<T_ * 32 * 32, 256, 0, stream>>>(ep, pwq, bq, tmp);
  inorm_lrelu<<<T_ * C_, 256, 0, stream>>>(tmp, qh, nullptr, 0);
  // k = ConvBlock(c+pos, Wk)
  conv3d_wmma<<<T_ * 32 * 32, 256, 0, stream>>>(cp, pwk, bk, tmp);
  inorm_lrelu<<<T_ * C_, 256, 0, stream>>>(tmp, kh, nullptr, 0);
  // v = ConvBlock(c+pos, Wv)
  conv3d_wmma<<<T_ * 32 * 32, 256, 0, stream>>>(cp, pwv, bv, tmp);
  inorm_lrelu<<<T_ * C_, 256, 0, stream>>>(tmp, vh, nullptr, 0);

  // a = W1 q + b1 ; bkk = W2 k + b2
  chan_gemm<<<dim3((T_ * S_) / 32, 2), 256, 0, stream>>>(qh, kh, pw1, pw2, b1, b2, ah, bkh);

  // fused attention -> agg + q (f16)
  attn_fuse<<<dim3(S_ / 32, T_), 256, 0, stream>>>(ah, bkh, vh, qh, pw3, b3, gq);

  // out = ConvBlock(agg + q, Wf) -> f32 channel-first
  conv3d_wmma<<<T_ * 32 * 32, 256, 0, stream>>>(gq, pwf, bf, tmp);
  inorm_lrelu<<<T_ * C_, 256, 0, stream>>>(tmp, nullptr, out, 1);
}
